// MyGPT2ModelMultiHeadAttention_2963527434268
// MI455X (gfx1250) — matrix operand order repaired and checked by reading, not compiled, not run on hardware
//
#include <hip/hip_runtime.h>

#define SEQ   2048
#define DIM   2048
#define TDIM  6144   // 3*DIM
#define NHEAD 16
#define HD    128

typedef __attribute__((ext_vector_type(16))) __bf16          v16bf;
typedef __attribute__((ext_vector_type(8)))  float           v8f;
typedef __attribute__((ext_vector_type(8)))  unsigned short  v8us;

union Frag16 { v16bf v; v8us h[2]; };

__device__ __forceinline__ unsigned short f2bf(float f) {
  unsigned int u = __float_as_uint(f);
  return (unsigned short)((u + 0x7FFFu + ((u >> 16) & 1u)) >> 16);  // RNE
}

// A-matrix 16x32 bf16 fragment: lanes 0-15 hold K {0..7, 16..23}, lanes 16-31 hold K {8..15, 24..31}
__device__ __forceinline__ v16bf load_afrag(const unsigned short* base, int kh) {
  Frag16 f;
  f.h[0] = *(const v8us*)(base + kh * 8);
  f.h[1] = *(const v8us*)(base + 16 + kh * 8);
  return f.v;
}
// B-matrix 32x16 bf16 fragment: lane holds column n=lane%16, lanes 0-15 hold K {0..7, 16..23}, lanes 16-31 hold K {8..15, 24..31}
__device__ __forceinline__ v16bf load_bfrag(const unsigned short* base, int kh) {
  Frag16 f;
  f.h[0] = *(const v8us*)(base + kh * 8);
  f.h[1] = *(const v8us*)(base + 16 + kh * 8);
  return f.v;
}

// ---------------- elementwise f32 -> bf16 ----------------
__global__ void cvt_bf16(const float* __restrict__ in, unsigned short* __restrict__ out, int n) {
  int i = blockIdx.x * blockDim.x + threadIdx.x;
  if (i < n) out[i] = f2bf(in[i]);
}

// ---------------- transpose + convert: W[K,N] f32 -> Wt[N,K] bf16 ----------------
__global__ void __launch_bounds__(256)
transpose_cvt(const float* __restrict__ W, unsigned short* __restrict__ Wt, int K, int N) {
  __shared__ float tile[32][33];
  const int nb = N >> 5;
  const int bx = blockIdx.x % nb;   // N tile
  const int by = blockIdx.x / nb;   // K tile
  const int tx = threadIdx.x & 31;
  const int ty = threadIdx.x >> 5;  // 0..7
  const int n  = bx * 32 + tx;
#pragma unroll
  for (int r = ty; r < 32; r += 8)
    tile[r][tx] = W[(size_t)(by * 32 + r) * N + n];
  __syncthreads();
  const int k = by * 32 + tx;
#pragma unroll
  for (int r = ty; r < 32; r += 8)
    Wt[(size_t)(bx * 32 + r) * K + k] = f2bf(tile[tx][r]);
}

// ---------------- bf16 GEMM: C[M,N] = A[M,K] x Bt[N,K]^T + bias ----------------
// Workgroup = 8 waves -> tile M=128, N=64. B panel (64x32 bf16 = 4KB) is staged
// into LDS once per K-step and shared by all 8 waves (double-buffered, one
// barrier per step). A fragments stream from global (two 16B chunks per lane).
template <bool F32OUT>
__global__ void __launch_bounds__(256)
gemm_bf16(const unsigned short* __restrict__ A, const unsigned short* __restrict__ Bt,
          const float* __restrict__ bias, void* __restrict__ Cout,
          int M, int N, int K) {
  const int tid  = threadIdx.x;
  const int lane = tid & 31;
  const int wave = tid >> 5;
  const int nt64 = N >> 6;
  const int mt = blockIdx.x / nt64;
  const int nt = blockIdx.x % nt64;
  const int m0 = mt * 128 + wave * 16;
  const int n0 = nt * 64;
  const int rl = lane & 15, kh = lane >> 4;

  __shared__ alignas(32) unsigned short Bpan[2][64][32];   // double-buffered B panel

  // staging assignment: thread -> (row, 8-half chunk)
  const int srow = tid >> 2;          // 0..63
  const int sc8  = (tid & 3) * 8;     // 0,8,16,24
  const unsigned short* bsrc = Bt + (size_t)(n0 + srow) * K + sc8;

  v8f acc[4] = {};
  const unsigned short* arow = A + (size_t)(m0 + rl) * K;
  const int nk = K >> 5;

  // prologue: stage panel 0
  {
    v8us st = *(const v8us*)bsrc;
    *(v8us*)&Bpan[0][srow][sc8] = st;
  }

  for (int kb = 0; kb < nk; ++kb) {
    __syncthreads();                       // staged panel kb visible; old reads done
    const bool have_next = (kb + 1) < nk;
    v8us nxt = {};
    if (have_next) nxt = *(const v8us*)(bsrc + (kb + 1) * 32);
    if (kb + 2 < nk) __builtin_prefetch(bsrc + (kb + 2) * 32, 0, 3);

    v16bf af = load_afrag(arow + kb * 32, kh);
    const unsigned short* bp = &Bpan[kb & 1][0][0];
#pragma unroll
    for (int s = 0; s < 4; ++s) {
      v16bf bf = load_bfrag(bp + (s * 16 + rl) * 32, kh);
      acc[s] = __builtin_amdgcn_wmma_f32_16x16x32_bf16(false, af, false, bf,
                                                       (short)0, acc[s], false, false);
    }
    if (have_next) *(v8us*)&Bpan[(kb + 1) & 1][srow][sc8] = nxt;
  }

#pragma unroll
  for (int s = 0; s < 4; ++s) {
    const int col = n0 + s * 16 + rl;
    const float bv = bias[col];
#pragma unroll
    for (int i = 0; i < 8; ++i) {
      const int row = m0 + i + kh * 8;
      float v = acc[s][i] + bv;
      if (F32OUT) ((float*)Cout)[(size_t)row * N + col] = v;
      else        ((unsigned short*)Cout)[(size_t)row * N + col] = f2bf(v);
    }
  }
}

// ---------------- flash attention: 4 waves/WG, 64 q-rows of one head ----------------
__global__ void __launch_bounds__(128)
attn_kernel(const unsigned short* __restrict__ qkv, unsigned short* __restrict__ oattn) {
  const int head = blockIdx.x / (SEQ / 64);
  const int qblk = blockIdx.x % (SEQ / 64);
  const int tid  = threadIdx.x;
  const int wave = tid >> 5;
  const int lane = tid & 31;
  const int rl   = lane & 15;
  const int kh   = lane >> 4;
  const int qbase = qblk * 64 + wave * 16;

  __shared__ alignas(32) unsigned short Vt[HD][32];        // V^T for current key block
  __shared__ alignas(32) float          Stile[4][16][32];  // per-wave score tile
  __shared__ alignas(32) unsigned short Ptile[4][16][32];  // per-wave probabilities
  __shared__ float mrow[4][16], lrow[4][16], arow[4][16];

  // Q fragments (A-matrix, 4 chunks of K=32 over head_dim 128)
  const unsigned short* qrow = qkv + (size_t)(qbase + rl) * TDIM + head * HD;
  v16bf qf[4];
#pragma unroll
  for (int c = 0; c < 4; ++c) qf[c] = load_afrag(qrow + c * 32, kh);

  v8f oacc[8] = {};
  if (lane < 16) { mrow[wave][lane] = -3.0e38f; lrow[wave][lane] = 0.0f; }

  const int nkb = qblk * 2 + 2;                   // causal key-block limit
  for (int kb = 0; kb < nkb; ++kb) {
    const int kbase = kb * 32;
    __syncthreads();
    // cooperatively stage V^T (coalesced global reads, transposed LDS writes)
    for (int idx = tid; idx < 32 * HD; idx += 128) {
      int kk = idx >> 7, dd = idx & 127;
      Vt[dd][kk] = qkv[(size_t)(kbase + kk) * TDIM + 2 * DIM + head * HD + dd];
    }
    __syncthreads();

    // S = Q K^T : two 16-key tiles, each 4 chained WMMAs over head_dim
#pragma unroll
    for (int t = 0; t < 2; ++t) {
      v8f s = {};
#pragma unroll
      for (int c = 0; c < 4; ++c) {
        const unsigned short* krow =
            qkv + (size_t)(kbase + t * 16 + rl) * TDIM + DIM + head * HD + c * 32;
        v16bf kf = load_bfrag(krow, kh);
        s = __builtin_amdgcn_wmma_f32_16x16x32_bf16(false, qf[c], false, kf,
                                                    (short)0, s, false, false);
      }
#pragma unroll
      for (int i = 0; i < 8; ++i)
        Stile[wave][i + kh * 8][t * 16 + rl] = s[i];
    }

    // online softmax: one lane per query row
    if (lane < 16) {
      const int r = lane, grow = qbase + r;
      float m_old = mrow[wave][r];
      float bm = -3.0e38f;
      float sv[32];
#pragma unroll
      for (int c = 0; c < 32; ++c) {
        float s = Stile[wave][r][c] * 0.088388347648318447f;  // 1/sqrt(128)
        if (kbase + c > grow) s = -3.0e38f;                    // causal mask
        sv[c] = s;
        bm = fmaxf(bm, s);
      }
      float mnew  = fmaxf(m_old, bm);
      float alpha = __expf(m_old - mnew);
      float lsum  = 0.0f;
#pragma unroll
      for (int c = 0; c < 32; ++c) {
        float p = __expf(sv[c] - mnew);
        lsum += p;
        Ptile[wave][r][c] = f2bf(p);
      }
      mrow[wave][r] = mnew;
      lrow[wave][r] = lrow[wave][r] * alpha + lsum;
      arow[wave][r] = alpha;
    }

    // rescale running output, then O += P V
    float al[8];
#pragma unroll
    for (int i = 0; i < 8; ++i) al[i] = arow[wave][i + kh * 8];
#pragma unroll
    for (int ch = 0; ch < 8; ++ch)
#pragma unroll
      for (int i = 0; i < 8; ++i) oacc[ch][i] *= al[i];

    v16bf pf = load_afrag(&Ptile[wave][rl][0], kh);
#pragma unroll
    for (int ch = 0; ch < 8; ++ch) {
      v16bf vf = load_bfrag(&Vt[ch * 16 + rl][0], kh);
      oacc[ch] = __builtin_amdgcn_wmma_f32_16x16x32_bf16(false, pf, false, vf,
                                                         (short)0, oacc[ch], false, false);
    }
  }

  // epilogue: normalize by l and store bf16 into [S, D] (merged heads)
  float li[8];
#pragma unroll
  for (int i = 0; i < 8; ++i) li[i] = 1.0f / lrow[wave][i + kh * 8];
#pragma unroll
  for (int ch = 0; ch < 8; ++ch)
#pragma unroll
    for (int i = 0; i < 8; ++i) {
      int grow = qbase + i + kh * 8;
      int gcol = head * HD + ch * 16 + rl;
      oattn[(size_t)grow * DIM + gcol] = f2bf(oacc[ch][i] * li[i]);
    }
}

// ---------------- launch ----------------
extern "C" void kernel_launch(void* const* d_in, const int* in_sizes, int n_in,
                              void* d_out, int out_size, void* d_ws, size_t ws_size,
                              hipStream_t stream) {
  (void)in_sizes; (void)n_in; (void)out_size; (void)ws_size;
  const float* x      = (const float*)d_in[0];
  const float* W_attn = (const float*)d_in[1];
  const float* b_attn = (const float*)d_in[2];
  const float* W_proj = (const float*)d_in[3];
  const float* b_proj = (const float*)d_in[4];
  float* out = (float*)d_out;

  unsigned short* xh  = (unsigned short*)d_ws;                 // [S,D]  bf16
  unsigned short* wat = xh  + (size_t)SEQ * DIM;               // [3D,D] bf16 (W_attn^T)
  unsigned short* wpt = wat + (size_t)DIM * TDIM;              // [D,D]  bf16 (W_proj^T)
  unsigned short* qkv = wpt + (size_t)DIM * DIM;               // [S,3D] bf16
  unsigned short* oat = qkv + (size_t)SEQ * TDIM;              // [S,D]  bf16

  cvt_bf16<<<(SEQ * DIM + 255) / 256, 256, 0, stream>>>(x, xh, SEQ * DIM);
  transpose_cvt<<<(TDIM / 32) * (DIM / 32), 256, 0, stream>>>(W_attn, wat, DIM, TDIM);
  transpose_cvt<<<(DIM / 32) * (DIM / 32), 256, 0, stream>>>(W_proj, wpt, DIM, DIM);

  // qkv = x @ W_attn + b_attn  (bf16 out), M=2048 N=6144 K=2048
  gemm_bf16<false><<<(SEQ / 128) * (TDIM / 64), 256, 0, stream>>>(
      xh, wat, b_attn, qkv, SEQ, TDIM, DIM);

  attn_kernel<<<NHEAD * (SEQ / 64), 128, 0, stream>>>(qkv, oat);

  // out = attn_out @ W_proj + b_proj  (f32 out), M=2048 N=2048 K=2048
  gemm_bf16<true><<<(SEQ / 128) * (DIM / 64), 256, 0, stream>>>(
      oat, wpt, b_proj, out, SEQ, DIM, DIM);
}
